// BinaryLinear_24395414241518
// MI455X (gfx1250) — 8-bit matrix operand order repaired and checked by reading, compiled, not run on hardware
//
#include <hip/hip_runtime.h>


typedef int v8i  __attribute__((ext_vector_type(8)));
typedef int v4ig __attribute__((vector_size(16)));

#define BM 128
#define BN 128
#define BK 64

#define AS1 __attribute__((address_space(1)))
#define AS3 __attribute__((address_space(3)))

#define GPTR(p) ((AS1 v4ig*)(AS1 void*)(void*)(p))
#define LPTR(p) ((AS3 v4ig*)(AS3 void*)(void*)(p))

#if defined(__HIP_DEVICE_COMPILE__) &&                                         \
    __has_builtin(__builtin_amdgcn_global_load_async_to_lds_b128) &&           \
    __has_builtin(__builtin_amdgcn_s_wait_asynccnt)
#define HAVE_ASYNC_LDS 1
#else
#define HAVE_ASYNC_LDS 0
#endif

__global__ __launch_bounds__(256) void binarize_pm1_kernel(
    const float* __restrict__ src, const float* __restrict__ thr,
    signed char* __restrict__ dst, int n4, int Kdim) {
  int t = blockIdx.x * blockDim.x + threadIdx.x;
  if (t >= n4) return;
  float4 v = ((const float4*)src)[t];
  float th = 0.0f;
  if (thr) th = thr[((size_t)t * 4) / (size_t)Kdim];
  char4 r;
  r.x = (v.x >= th) ? (signed char)1 : (signed char)-1;
  r.y = (v.y >= th) ? (signed char)1 : (signed char)-1;
  r.z = (v.z >= th) ? (signed char)1 : (signed char)-1;
  r.w = (v.w >= th) ? (signed char)1 : (signed char)-1;
  ((char4*)dst)[t] = r;
}

__global__ __launch_bounds__(256) void bgemm_wmma_iu8_kernel(
    const signed char* __restrict__ A, const signed char* __restrict__ Bm,
    float* __restrict__ C, const float* __restrict__ shift_param,
    int M, int N, int K) {
  __shared__ signed char lA[2][BM * BK];
  __shared__ signed char lB[2][BN * BK];

  const int tid   = threadIdx.x;
  const int wave  = tid >> 5;
  const int lane  = tid & 31;
  const int wm    = wave >> 2;
  const int wn    = wave & 3;
  const int l16   = lane & 15;
  const int khalf = lane >> 4;

  const int m0 = blockIdx.y * BM;
  const int n0 = blockIdx.x * BN;

  union Frag { v8i v; int2 d2[4]; int4 d4[2]; };

  const v8i vzero = {};
  v8i acc[4][2];
#pragma unroll
  for (int mt = 0; mt < 4; ++mt)
#pragma unroll
    for (int nt = 0; nt < 2; ++nt) acc[mt][nt] = vzero;

  const int nk = K / BK;

#if HAVE_ASYNC_LDS
  auto issue_tile = [&](int k0, int buf) {
#pragma unroll
    for (int i = 0; i < 2; ++i) {
      int c = tid + i * 256;
      int row = c >> 2, col = (c & 3) * 16;
      __builtin_amdgcn_global_load_async_to_lds_b128(
          GPTR(A + (size_t)(m0 + row) * K + k0 + col),
          LPTR(lA[buf] + row * BK + col), 0, 0);
      __builtin_amdgcn_global_load_async_to_lds_b128(
          GPTR(Bm + (size_t)(n0 + row) * K + k0 + col),
          LPTR(lB[buf] + row * BK + col), 0, 0);
    }
  };

  issue_tile(0, 0);
#else
  auto fill_tile = [&](int k0, int buf) {
#pragma unroll
    for (int i = 0; i < 2; ++i) {
      int c = tid + i * 256;
      int row = c >> 2, col = (c & 3) * 16;
      int4 va = *(const int4*)(A  + (size_t)(m0 + row) * K + k0 + col);
      int4 vb = *(const int4*)(Bm + (size_t)(n0 + row) * K + k0 + col);
      *(int4*)(lA[buf] + row * BK + col) = va;
      *(int4*)(lB[buf] + row * BK + col) = vb;
    }
  };
#endif

  for (int ks = 0; ks < nk; ++ks) {
    const int buf = ks & 1;

#if HAVE_ASYNC_LDS
    if (ks + 1 < nk) {
      issue_tile((ks + 1) * BK, (ks + 1) & 1);
      __builtin_amdgcn_s_wait_asynccnt(4);
    } else {
      __builtin_amdgcn_s_wait_asynccnt(0);
    }
    __syncthreads();
#else
    fill_tile(ks * BK, buf);
    __syncthreads();
#endif

    Frag af[4], bf[2];
#pragma unroll
    for (int mt = 0; mt < 4; ++mt) {
      const int2* p = (const int2*)(lA[buf] + (wm * 64 + mt * 16 + l16) * BK);
      af[mt].d2[0] = p[0 + 2 * khalf];
      af[mt].d2[1] = p[1 + 2 * khalf];
      af[mt].d2[2] = p[4 + 2 * khalf];
      af[mt].d2[3] = p[5 + 2 * khalf];
    }
#pragma unroll
    for (int nt = 0; nt < 2; ++nt) {
      const int4* q = (const int4*)(lB[buf] + (wn * 32 + nt * 16 + l16) * BK);
      bf[nt].d4[0] = q[0 + khalf];
      bf[nt].d4[1] = q[2 + khalf];
    }

#pragma unroll
    for (int mt = 0; mt < 4; ++mt)
#pragma unroll
      for (int nt = 0; nt < 2; ++nt)
        acc[mt][nt] = __builtin_amdgcn_wmma_i32_16x16x64_iu8(
            true, af[mt].v, true, bf[nt].v, acc[mt][nt], false, false);

    __syncthreads();
  }

  float sp = shift_param[0];
  sp = fminf(fmaxf(sp, -8.0f), 0.0f);
  const float scale = exp2f(rintf(sp));

#pragma unroll
  for (int mt = 0; mt < 4; ++mt) {
#pragma unroll
    for (int nt = 0; nt < 2; ++nt) {
      const int col     = n0 + wn * 32 + nt * 16 + l16;
      const int rowbase = m0 + wm * 64 + mt * 16 + khalf * 8;
#pragma unroll
      for (int v = 0; v < 8; ++v)
        C[(size_t)(rowbase + v) * N + col] = (float)acc[mt][nt][v] * scale;
    }
  }
}

extern "C" void kernel_launch(void* const* d_in, const int* in_sizes, int n_in,
                              void* d_out, int out_size, void* d_ws, size_t ws_size,
                              hipStream_t stream) {
  const float* x     = (const float*)d_in[0];
  const float* w     = (const float*)d_in[1];
  const float* thr   = (const float*)d_in[2];
  const float* shift = (const float*)d_in[3];
  float*       out   = (float*)d_out;

  const int N = in_sizes[2];
  const int K = in_sizes[1] / N;
  const int M = in_sizes[0] / K;

  signed char* xq = (signed char*)d_ws;
  signed char* wq = xq + (size_t)M * (size_t)K;

  {
    int n4 = (M * K) / 4;
    binarize_pm1_kernel<<<(n4 + 255) / 256, 256, 0, stream>>>(x, nullptr, xq, n4, K);
  }
  {
    int n4 = (N * K) / 4;
    binarize_pm1_kernel<<<(n4 + 255) / 256, 256, 0, stream>>>(w, thr, wq, n4, K);
  }

  dim3 grid(N / BN, M / BM);
  bgemm_wmma_iu8_kernel<<<grid, 256, 0, stream>>>(xq, wq, out, shift, M, N, K);
}
